// DeMaBlock_73598559584850
// MI455X (gfx1250) — matrix operand order repaired and checked by reading, not compiled, not run on hardware
//
#include <hip/hip_runtime.h>
#include <stdint.h>

// ---------------- problem constants ----------------
constexpr int kB  = 4;
constexpr int kL  = 2048;
constexpr int kDM = 1024;            // d_model
constexpr int kDI = 2048;            // d_inner
constexpr int kDS = 16;              // d_state
constexpr int kDR = 64;              // dt_rank
constexpr int kM  = kB * kL;         // 8192 tokens
constexpr int kE  = 2 * kDI;         // 4096 (xz width)
constexpr int kXD = kDR + 2 * kDS;   // 96  (x_dbl width)

// ---------------- types ----------------
typedef __attribute__((ext_vector_type(16))) __bf16 v16bf;
typedef __attribute__((ext_vector_type(8)))  float  v8f;
typedef int v4i_vs __attribute__((vector_size(16)));   // matches builtin V4i

union Frag {            // 32 bytes: one WMMA 16-bit operand per lane
  uint4 u[2];
  v16bf v;
};

#define GLOBAL_AS __attribute__((address_space(1)))
#define LDS_AS    __attribute__((address_space(3)))

// Probe the CDNA5 async global->LDS path (ASYNCcnt). Host pass sees 0.
#if defined(__gfx1250__) && __has_builtin(__builtin_amdgcn_global_load_async_to_lds_b128)
#define HAVE_ASYNC 1
#else
#define HAVE_ASYNC 0
#endif

#if HAVE_ASYNC && __has_builtin(__builtin_amdgcn_s_wait_asynccnt)
#define WAIT_ASYNC(n) __builtin_amdgcn_s_wait_asynccnt((n))
#elif HAVE_ASYNC
#define WAIT_ASYNC(n) asm volatile("s_wait_asynccnt %0" ::"i"(n) : "memory")
#else
#define WAIT_ASYNC(n) ((void)0)
#endif

__device__ __forceinline__ void copy16_g2l(const uint16_t* g, uint16_t* l) {
#if HAVE_ASYNC
  __builtin_amdgcn_global_load_async_to_lds_b128(
      (GLOBAL_AS v4i_vs*)g, (LDS_AS v4i_vs*)l, 0, 0);
#else
  *(uint4*)l = *(const uint4*)g;
#endif
}

__device__ __forceinline__ uint16_t f2bf(float f) {
  uint32_t u = __float_as_uint(f);
  uint32_t r = u + 0x7FFFu + ((u >> 16) & 1u);   // round-to-nearest-even
  return (uint16_t)(r >> 16);
}

__device__ __forceinline__ float softplus_f(float x) {
  return (x > 20.f) ? x : log1pf(__expf(x));
}

__device__ __forceinline__ float sigmoid_f(float x) {
  return 1.f / (1.f + __expf(-x));
}

// ---------------- elementwise converts ----------------
__global__ void f32_to_bf16_kernel(const float* __restrict__ src,
                                   uint16_t* __restrict__ dst, int n) {
  int i = blockIdx.x * blockDim.x + threadIdx.x;
  if (i < n) dst[i] = f2bf(src[i]);
}

// dt slice (first 64 of 96 cols of x_dbl) -> bf16
__global__ void dt_slice_bf16_kernel(const float* __restrict__ xdbl,
                                     uint16_t* __restrict__ dt) {
  int i = blockIdx.x * blockDim.x + threadIdx.x;
  if (i < kM * kDR) {
    int m = i >> 6, k = i & 63;
    dt[i] = f2bf(xdbl[(size_t)m * kXD + k]);
  }
}

// ---------------- WMMA GEMM: C[M,N] = A[M,K] * W[N,K]^T (bf16 in, f32 out) --
// Double-buffered BK=64 stages staged via async global->LDS copies.
// EPI: 0 = plain store, 1 = softplus(acc + bias[n]) (for delta)
template <int BM, int BN, int BK, int WGM, int WGN, int WM, int WN, int EPI,
          int N, int K>
__global__ __launch_bounds__(32 * WGM * WGN)
void gemm_bf16_wmma(const uint16_t* __restrict__ A,
                    const uint16_t* __restrict__ W,
                    float* __restrict__ C,
                    const float* __restrict__ bias) {
  constexpr int NTHREADS = 32 * WGM * WGN;
  constexpr int CPR = BK / 8;                       // 16B chunks per row
  constexpr int LA  = (BM * CPR) / NTHREADS;        // A copies per thread/stage
  constexpr int LB  = (BN * CPR) / NTHREADS;        // W copies per thread/stage
  constexpr int NSTAGE = LA + LB;                   // async ops per stage
  constexpr int RSTEP  = NTHREADS / CPR;            // rows covered per pass

  __shared__ uint16_t As[2][BM][BK];
  __shared__ uint16_t Ws[2][BN][BK];

  const int tid  = threadIdx.x;
  const int lane = tid & 31;
  const int wave = tid >> 5;
  const int wm   = wave / WGN;
  const int wn   = wave % WGN;
  const int m0   = blockIdx.y * BM;
  const int n0   = blockIdx.x * BN;

  // hoisted per-thread copy coordinates
  const int rT = tid / CPR;             // tile row
  const int cT = (tid % CPR) * 8;       // element col within BK
  const uint16_t* gA = A + (size_t)(m0 + rT) * K + cT;
  const uint16_t* gW = W + (size_t)(n0 + rT) * K + cT;

  auto issue = [&](int k0, int buf) {
#pragma unroll
    for (int q = 0; q < LA; ++q)
      copy16_g2l(gA + (size_t)q * RSTEP * K + k0, &As[buf][rT + q * RSTEP][cT]);
#pragma unroll
    for (int q = 0; q < LB; ++q)
      copy16_g2l(gW + (size_t)q * RSTEP * K + k0, &Ws[buf][rT + q * RSTEP][cT]);
  };

  v8f acc[WM][WN];
#pragma unroll
  for (int i = 0; i < WM; ++i)
#pragma unroll
    for (int j = 0; j < WN; ++j)
      acc[i][j] = (v8f){0.f, 0.f, 0.f, 0.f, 0.f, 0.f, 0.f, 0.f};

  const int mr = lane & 15;
  const int hi = lane >> 4;

  issue(0, 0);
  int cur = 0;
  for (int k0 = 0; k0 < K; k0 += BK) {
    if (k0 + BK < K) {
      issue(k0 + BK, cur ^ 1);
      if (k0 + 2 * BK < K) {        // L2 prefetch of stage i+2
        __builtin_prefetch(gA + k0 + 2 * BK, 0, 1);
        __builtin_prefetch(gW + k0 + 2 * BK, 0, 1);
      }
      WAIT_ASYNC(NSTAGE);           // stage i done; stage i+1 still in flight
    } else {
      WAIT_ASYNC(0);
    }
    __syncthreads();

#pragma unroll
    for (int ks = 0; ks < BK; ks += 32) {
      Frag a[WM], b[WN];
#pragma unroll
      for (int i = 0; i < WM; ++i) {
        int row = wm * WM * 16 + i * 16 + mr;
        a[i].u[0] = *(const uint4*)&As[cur][row][ks + hi * 8];
        a[i].u[1] = *(const uint4*)&As[cur][row][ks + hi * 8 + 16];
      }
#pragma unroll
      for (int j = 0; j < WN; ++j) {
        int row = wn * WN * 16 + j * 16 + mr;
        b[j].u[0] = *(const uint4*)&Ws[cur][row][ks + hi * 8];
        b[j].u[1] = *(const uint4*)&Ws[cur][row][ks + hi * 8 + 16];
      }
#pragma unroll
      for (int i = 0; i < WM; ++i)
#pragma unroll
        for (int j = 0; j < WN; ++j)
          acc[i][j] = __builtin_amdgcn_wmma_f32_16x16x32_bf16(
              false, a[i].v, false, b[j].v, (short)0, acc[i][j], false, false);
    }
    __syncthreads();   // stage i fully consumed before buffer reuse
    cur ^= 1;
  }

  // epilogue: lane l -> col = l&15; VGPR r -> row = 8*(l>>4) + r
#pragma unroll
  for (int i = 0; i < WM; ++i)
#pragma unroll
    for (int j = 0; j < WN; ++j) {
      int rbase = m0 + wm * WM * 16 + i * 16 + hi * 8;
      int col   = n0 + wn * WN * 16 + j * 16 + mr;
      float bv  = (EPI == 1) ? bias[col] : 0.f;
#pragma unroll
      for (int r = 0; r < 8; ++r) {
        float v = acc[i][j][r];
        if (EPI == 1) v = softplus_f(v + bv);
        C[(size_t)(rbase + r) * N + col] = v;
      }
    }
}

// ---------------- depthwise causal conv (K=4) + bias + SiLU ----------------
__global__ void conv_silu_kernel(const float* __restrict__ xz,
                                 const float* __restrict__ conv_w,
                                 const float* __restrict__ conv_b,
                                 float* __restrict__ u_f32,
                                 uint16_t* __restrict__ u_bf16) {
  int idx = blockIdx.x * blockDim.x + threadIdx.x;
  if (idx >= kM * kDI) return;
  int d  = idx % kDI;
  int ml = idx / kDI;           // b*L + l
  int l  = ml % kL;
  float acc = conv_b[d];
#pragma unroll
  for (int j = 0; j < 4; ++j) {
    int ll = l - 3 + j;
    if (ll >= 0)
      acc += conv_w[d * 4 + j] * xz[(size_t)(ml - 3 + j) * kE + d];
  }
  float u = acc * sigmoid_f(acc);
  u_f32[idx]  = u;
  u_bf16[idx] = f2bf(u);
}

// ---------------- selective scan (serial over L, parallel over b*d) --------
__global__ __launch_bounds__(256)
void scan_kernel(const float* __restrict__ delta,
                 const float* __restrict__ xdbl,   // [M,96]: B at 64, C at 80
                 const float* __restrict__ u,
                 const float* __restrict__ xz,     // z at [..., 2048+d]
                 const float* __restrict__ A_log,
                 const float* __restrict__ D_param,
                 uint16_t* __restrict__ y_bf16) {
  const int tid = threadIdx.x;
  const int b   = blockIdx.x >> 3;
  const int d   = (blockIdx.x & 7) * 256 + tid;

  float a[kDS], h[kDS];
#pragma unroll
  for (int n = 0; n < kDS; ++n) {
    a[n] = -__expf(A_log[d * kDS + n]);
    h[n] = 0.f;
  }
  const float Dv = D_param[d];

  constexpr int CH = 64;              // timesteps staged per LDS chunk
  __shared__ float BC[CH][32];        // [s][0:16]=B_t, [s][16:32]=C_t

  for (int l0 = 0; l0 < kL; l0 += CH) {
    __syncthreads();
    for (int i = tid; i < CH * 32; i += 256) {
      int s = i >> 5, j = i & 31;
      BC[s][j] = xdbl[(size_t)(b * kL + l0 + s) * kXD + 64 + j];
    }
    __syncthreads();

    for (int s = 0; s < CH; ++s) {
      const size_t idx = (size_t)(b * kL + l0 + s) * kDI + d;
      const float dv = delta[idx];
      const float uv = u[idx];
      float y = 0.f;
#pragma unroll
      for (int n = 0; n < kDS; ++n) {
        float dA = __expf(dv * a[n]);
        h[n] = dA * h[n] + dv * BC[s][n] * uv;
        y += h[n] * BC[s][16 + n];
      }
      const float zv = xz[(size_t)(b * kL + l0 + s) * kE + kDI + d];
      const float yo = (y + uv * Dv) * (zv * sigmoid_f(zv));
      y_bf16[idx] = f2bf(yo);
    }
  }
}

// ---------------- residual + LayerNorm ----------------
__global__ __launch_bounds__(256)
void layernorm_kernel(const float* __restrict__ outm,
                      const float* __restrict__ x,
                      const float* __restrict__ w,
                      const float* __restrict__ bln,
                      float* __restrict__ out) {
  const int row = blockIdx.x;
  const int tid = threadIdx.x;
  float v[4], s = 0.f, s2 = 0.f;
#pragma unroll
  for (int i = 0; i < 4; ++i) {
    int c = tid + i * 256;
    float t = outm[(size_t)row * kDM + c] + x[(size_t)row * kDM + c];
    v[i] = t; s += t; s2 += t * t;
  }
  __shared__ float r1[256], r2[256];
  r1[tid] = s; r2[tid] = s2;
  __syncthreads();
  for (int stp = 128; stp > 0; stp >>= 1) {
    if (tid < stp) { r1[tid] += r1[tid + stp]; r2[tid] += r2[tid + stp]; }
    __syncthreads();
  }
  const float mu  = r1[0] * (1.f / kDM);
  const float var = r2[0] * (1.f / kDM) - mu * mu;
  const float rs  = rsqrtf(var + 1e-5f);
#pragma unroll
  for (int i = 0; i < 4; ++i) {
    int c = tid + i * 256;
    out[(size_t)row * kDM + c] = (v[i] - mu) * rs * w[c] + bln[c];
  }
}

// ---------------- host orchestration ----------------
extern "C" void kernel_launch(void* const* d_in, const int* in_sizes, int n_in,
                              void* d_out, int out_size, void* d_ws, size_t ws_size,
                              hipStream_t stream) {
  const float* x       = (const float*)d_in[0];
  const float* W_in    = (const float*)d_in[1];
  const float* conv_w  = (const float*)d_in[2];
  const float* conv_b  = (const float*)d_in[3];
  const float* W_x     = (const float*)d_in[4];
  const float* W_dt    = (const float*)d_in[5];
  const float* b_dt    = (const float*)d_in[6];
  const float* A_log   = (const float*)d_in[7];
  const float* D_param = (const float*)d_in[8];
  const float* W_out   = (const float*)d_in[9];
  const float* ln_w    = (const float*)d_in[10];
  const float* ln_b    = (const float*)d_in[11];
  float* out = (float*)d_out;

  // workspace carve-out
  size_t off = 0;
  auto take = [&](size_t bytes) -> char* {
    char* p = (char*)d_ws + off;
    off += (bytes + 255) & ~(size_t)255;
    return p;
  };
  float*    xz      = (float*)   take((size_t)kM * kE  * 4);
  float*    u_f32   = (float*)   take((size_t)kM * kDI * 4);
  uint16_t* u_bf16  = (uint16_t*)take((size_t)kM * kDI * 2);
  float*    xdbl    = (float*)   take((size_t)kM * kXD * 4);
  uint16_t* dt_bf   = (uint16_t*)take((size_t)kM * kDR * 2);
  float*    delta   = (float*)   take((size_t)kM * kDI * 4);
  uint16_t* y_bf    = (uint16_t*)take((size_t)kM * kDI * 2);
  float*    outm    = (float*)   take((size_t)kM * kDM * 4);
  uint16_t* x_bf    = (uint16_t*)take((size_t)kM * kDM * 2);
  uint16_t* win_bf  = (uint16_t*)take((size_t)kE  * kDM * 2);
  uint16_t* wx_bf   = (uint16_t*)take((size_t)kXD * kDI * 2);
  uint16_t* wdt_bf  = (uint16_t*)take((size_t)kDI * kDR * 2);
  uint16_t* wout_bf = (uint16_t*)take((size_t)kDM * kDI * 2);

  auto cvt = [&](const float* s, uint16_t* dst, int n) {
    f32_to_bf16_kernel<<<(n + 255) / 256, 256, 0, stream>>>(s, dst, n);
  };
  cvt(x,     x_bf,    kM  * kDM);
  cvt(W_in,  win_bf,  kE  * kDM);
  cvt(W_x,   wx_bf,   kXD * kDI);
  cvt(W_dt,  wdt_bf,  kDI * kDR);
  cvt(W_out, wout_bf, kDM * kDI);

  // GEMM1: xz = x @ W_in^T   (8192 x 4096 x 1024)
  gemm_bf16_wmma<128, 128, 64, 4, 2, 2, 4, 0, kE, kDM>
      <<<dim3(kE / 128, kM / 128), 256, 0, stream>>>(x_bf, win_bf, xz, nullptr);

  // depthwise conv + SiLU -> u
  conv_silu_kernel<<<(kM * kDI + 255) / 256, 256, 0, stream>>>(
      xz, conv_w, conv_b, u_f32, u_bf16);

  // GEMM2: x_dbl = u @ W_x^T  (8192 x 96 x 2048), narrow-N config
  gemm_bf16_wmma<128, 32, 64, 8, 1, 1, 2, 0, kXD, kDI>
      <<<dim3(kXD / 32, kM / 128), 256, 0, stream>>>(u_bf16, wx_bf, xdbl, nullptr);

  // dt slice -> bf16
  dt_slice_bf16_kernel<<<(kM * kDR + 255) / 256, 256, 0, stream>>>(xdbl, dt_bf);

  // GEMM3: delta = softplus(dt @ W_dt^T + b_dt)  (8192 x 2048 x 64)
  gemm_bf16_wmma<128, 128, 64, 4, 2, 2, 4, 1, kDI, kDR>
      <<<dim3(kDI / 128, kM / 128), 256, 0, stream>>>(dt_bf, wdt_bf, delta, b_dt);

  // selective scan + gating, fused, -> y (bf16)
  scan_kernel<<<kB * (kDI / 256), 256, 0, stream>>>(
      delta, xdbl, u_f32, xz, A_log, D_param, y_bf);

  // GEMM4: outm = y @ W_out^T  (8192 x 1024 x 2048)
  gemm_bf16_wmma<128, 128, 64, 4, 2, 2, 4, 0, kDM, kDI>
      <<<dim3(kDM / 128, kM / 128), 256, 0, stream>>>(y_bf, wout_bf, outm, nullptr);

  // residual + layernorm -> d_out
  layernorm_kernel<<<kM, 256, 0, stream>>>(outm, x, ln_w, ln_b, out);

  (void)in_sizes; (void)n_in; (void)out_size; (void)ws_size;
}
